// GCIR3_v1_13649406067124
// MI455X (gfx1250) — compile-verified, matrix operand order checked by reading, not run on hardware
//
#include <hip/hip_runtime.h>
#include <hip/hip_bf16.h>
#include <math.h>

typedef __attribute__((ext_vector_type(16))) _Float16 v16h;
typedef __attribute__((ext_vector_type(8)))  float    v8f;

// ---------------- problem dims ----------------
#define NH   114          // rows (graph nodes)
#define W0   130
#define K1   65
#define W1   66           // 130-65+1
#define C1   5
#define K2   30
#define W2   37           // 66-30+1
#define C2   10
#define K3   37
#define C3   20           // feature dim
#define NPAD 128          // 114 padded to 128 for WMMA
#define KPAD 32           // 20 padded to 32 for WMMA

__device__ __forceinline__ float eluf(float v) {
    return v > 0.0f ? v : expm1f(v);
}

// ============================================================
// Kernel 1: fused conv1/LN/ELU -> conv2/LN/ELU -> conv3/LN/ELU
// One block per (batch, row). All 1xK convs => rows independent.
// ============================================================
__global__ void __launch_bounds__(64)
gcir3_conv_stack(const float* __restrict__ x,
                 const float* __restrict__ c1w, const float* __restrict__ c1b,
                 const float* __restrict__ l1g, const float* __restrict__ l1b,
                 const float* __restrict__ c2w, const float* __restrict__ c2b,
                 const float* __restrict__ l2g, const float* __restrict__ l2b,
                 const float* __restrict__ c3w, const float* __restrict__ c3b,
                 const float* __restrict__ l3g, const float* __restrict__ l3b,
                 float* __restrict__ feats)
{
    __shared__ float sx[W0];
    __shared__ float w1s[K1 * C1];          // 325
    __shared__ float w2s[K2 * C1 * C2];     // 1500
    __shared__ float w3s[K3 * C2 * C3];     // 7400
    __shared__ float h1s[W1 * C1];          // 330
    __shared__ float h2s[W2 * C2];          // 370
    __shared__ float c3s[C3];               // 20

    const int tid = threadIdx.x;
    const int br  = blockIdx.x;
    const int b   = br / NH;
    const int r   = br % NH;

    const float* xrow = x + ((size_t)b * NH + r) * W0;
    __builtin_prefetch(xrow, 0, 1);   // global_prefetch_b8

    for (int i = tid; i < W0; i += 64)             sx[i]  = xrow[i];
    for (int i = tid; i < K1 * C1; i += 64)        w1s[i] = c1w[i];
    for (int i = tid; i < K2 * C1 * C2; i += 64)   w2s[i] = c2w[i];
    for (int i = tid; i < K3 * C2 * C3; i += 64)   w3s[i] = c3w[i];
    __syncthreads();

    // ---- conv1: [130] -> [66][5] ----
    for (int idx = tid; idx < W1 * C1; idx += 64) {
        int ox = idx / C1, oc = idx % C1;
        float acc = c1b[oc];
        #pragma unroll 5
        for (int k = 0; k < K1; ++k) acc += sx[ox + k] * w1s[k * C1 + oc];
        h1s[idx] = acc;
    }
    __syncthreads();
    // ---- LN1 (over 5 channels) + ELU, in place ----
    if (tid < W1) {
        float m = 0.f;
        #pragma unroll
        for (int c = 0; c < C1; ++c) m += h1s[tid * C1 + c];
        m *= (1.0f / C1);
        float v = 0.f;
        #pragma unroll
        for (int c = 0; c < C1; ++c) { float d = h1s[tid * C1 + c] - m; v += d * d; }
        v *= (1.0f / C1);
        float inv = rsqrtf(v + 1e-3f);
        #pragma unroll
        for (int c = 0; c < C1; ++c) {
            float d = (h1s[tid * C1 + c] - m) * inv * l1g[c] + l1b[c];
            h1s[tid * C1 + c] = eluf(d);
        }
    }
    __syncthreads();

    // ---- conv2: [66][5] -> [37][10] ----
    for (int idx = tid; idx < W2 * C2; idx += 64) {
        int ox = idx / C2, oc = idx % C2;
        float acc = c2b[oc];
        for (int k = 0; k < K2; ++k)
            #pragma unroll
            for (int ic = 0; ic < C1; ++ic)
                acc += h1s[(ox + k) * C1 + ic] * w2s[(k * C1 + ic) * C2 + oc];
        h2s[idx] = acc;
    }
    __syncthreads();
    // ---- LN2 + ELU ----
    if (tid < W2) {
        float m = 0.f;
        #pragma unroll
        for (int c = 0; c < C2; ++c) m += h2s[tid * C2 + c];
        m *= (1.0f / C2);
        float v = 0.f;
        #pragma unroll
        for (int c = 0; c < C2; ++c) { float d = h2s[tid * C2 + c] - m; v += d * d; }
        v *= (1.0f / C2);
        float inv = rsqrtf(v + 1e-3f);
        #pragma unroll
        for (int c = 0; c < C2; ++c) {
            float d = (h2s[tid * C2 + c] - m) * inv * l2g[c] + l2b[c];
            h2s[tid * C2 + c] = eluf(d);
        }
    }
    __syncthreads();

    // ---- conv3: [37][10] -> [1][20] ----
    if (tid < C3) {
        float acc = c3b[tid];
        for (int k = 0; k < K3; ++k)
            #pragma unroll
            for (int ic = 0; ic < C2; ++ic)
                acc += h2s[k * C2 + ic] * w3s[(k * C2 + ic) * C3 + tid];
        c3s[tid] = acc;
    }
    __syncthreads();
    // ---- LN3 + ELU -> feats ----
    if (tid < C3) {
        float m = 0.f;
        #pragma unroll
        for (int c = 0; c < C3; ++c) m += c3s[c];
        m *= (1.0f / C3);
        float v = 0.f;
        #pragma unroll
        for (int c = 0; c < C3; ++c) { float d = c3s[c] - m; v += d * d; }
        v *= (1.0f / C3);
        float inv = rsqrtf(v + 1e-3f);
        float d = (c3s[tid] - m) * inv * l3g[tid] + l3b[tid];
        feats[((size_t)b * NH + r) * C3 + tid] = eluf(d);
    }
}

// ---------------- WMMA fragment helper (wave32) ----------------
// 16-bit A (16x32) fragment from a row-major tile, leading dim lda.
//   lanes 0-15  -> row M=lane,     K in {kb..kb+7, 16+kb..23+kb} kb=0
//   lanes 16-31 -> row M=lane-16,  kb=8
// The SAME pattern serves the B matrix (32x16) when B is stored
// transposed (BT[n][k] row-major): lane -> column n of B.
__device__ __forceinline__ v16h frag_rowmajor(const _Float16* base, int lda, int lane)
{
    int row = lane & 15;
    int kb  = (lane & 16) ? 8 : 0;
    const _Float16* p = base + row * lda + kb;
    v16h a;
    #pragma unroll
    for (int j = 0; j < 8; ++j) a[j]     = p[j];
    #pragma unroll
    for (int j = 0; j < 8; ++j) a[8 + j] = p[16 + j];
    return a;
}

// ============================================================
// Kernel 2: per-batch graph head. 256 threads = 8 wave32.
// feats -> index -> corr (WMMA) -> Laplacian (LDS f16) ->
// two L@Y graph convs (WMMA, RHS stored transposed) ->
// pooling -> dense -> softmax.
// ============================================================
__global__ void __launch_bounds__(256)
gcir3_graph_head(const float* __restrict__ feats,
                 const float* __restrict__ agw, const float* __restrict__ agb,
                 const float* __restrict__ g1w, const float* __restrict__ g1b,
                 const float* __restrict__ g2w, const float* __restrict__ g2b,
                 const float* __restrict__ dw,  const float* __restrict__ db,
                 const int*   __restrict__ rlp,
                 float* __restrict__ out)
{
    __shared__ float    sF[NH * C3];          // 9120 B  feats for this batch
    __shared__ float    sIdx[NPAD];           // agent gate
    __shared__ float    sInv[NPAD];           // 1/||fc|| then reused as d^-1/2
    __shared__ float    sScr[NPAD * 16];      // 8192 B: fcH (f16 [128][32]) then h2 (f32 [128][16])
    __shared__ _Float16 sL[NPAD * NPAD];      // 32768 B: dist -> A -> L (f16)
    __shared__ _Float16 sYT[16 * NPAD];       // RHS, TRANSPOSED: [n][k] (4096 B)
    __shared__ _Float16 sZT[16 * NPAD];       // h1, TRANSPOSED:  [n][m] (4096 B)
    __shared__ float    sRed[256];
    __shared__ float    sSigma;

    const int tid  = threadIdx.x;
    const int lane = tid & 31;
    const int wv   = tid >> 5;          // 0..7 -> M-tile
    const int b    = blockIdx.x;
    const int rl   = *rlp;

    _Float16* fcH = (_Float16*)sScr;    // [128][32] centered feats, f16
    float*    h2f = sScr;               // [128][16] reused later

    // ---- load feats[b] ----
    const float* fb = feats + (size_t)b * NH * C3;
    for (int i = tid; i < NH * C3; i += 256) sF[i] = fb[i];
    __syncthreads();

    // ---- agent index, centering, 1/norm, f16 staging ----
    if (tid < NPAD) {
        int i = tid;
        if (i < NH) {
            float m = 0.f;
            #pragma unroll
            for (int k = 0; k < C3; ++k) m += sF[i * C3 + k];
            m *= (1.0f / C3);
            float acc = agb[0], ss = 0.f;
            #pragma unroll
            for (int k = 0; k < C3; ++k) {
                float f  = sF[i * C3 + k];
                acc     += f * agw[k];
                float fc = f - m;
                ss      += fc * fc;
                fcH[i * KPAD + k] = (_Float16)fc;
            }
            #pragma unroll
            for (int k = C3; k < KPAD; ++k) fcH[i * KPAD + k] = (_Float16)0.f;
            sInv[i] = rsqrtf(ss);                   // 1/||fc||
            sIdx[i] = rl ? (1.0f / (1.0f + __expf(-acc))) : 1.0f;
        } else {
            #pragma unroll
            for (int k = 0; k < KPAD; ++k) fcH[i * KPAD + k] = (_Float16)0.f;
            sInv[i] = 0.0f;
            sIdx[i] = 0.0f;
        }
    }
    __syncthreads();

    // ---- corr = fc @ fc^T via WMMA (M=N=128, K=32) -> dist in sL ----
    // B[k][n] = fc[n][k]  ==  row-major read of fcH at row n  -> same
    // fragment pattern as A (contiguous b128 LDS loads).
    const int m0 = wv * 16;
    const v16h afrag = frag_rowmajor(fcH + m0 * KPAD, KPAD, lane);
    float localSum = 0.0f;
    for (int nt = 0; nt < 8; ++nt) {
        const int n0 = nt * 16;
        const v16h bfrag = frag_rowmajor(fcH + n0 * KPAD, KPAD, lane);

        v8f c = {};
        c = __builtin_amdgcn_wmma_f32_16x16x32_f16(false, afrag, false, bfrag,
                                                   (short)0, c, false, false);
        const int n  = n0 + (lane & 15);
        const int mb = m0 + ((lane & 16) ? 8 : 0);
        const float invN = sInv[n];
        #pragma unroll
        for (int rr = 0; rr < 8; ++rr) {
            const int m = mb + rr;
            float corr = c[rr] * (sInv[m] * invN);
            float dist = (m < NH && n < NH && m != n) ? (1.0f - corr) : 0.0f;
            localSum += dist;
            sL[m * NPAD + n] = (_Float16)dist;
        }
    }
    sRed[tid] = localSum;
    __syncthreads();
    if (tid == 0) {
        float s = 0.f;
        for (int i = 0; i < 256; ++i) s += sRed[i];
        sSigma = s / (float)(NH * NH);
    }
    __syncthreads();

    // ---- A = exp(-dist^2/(2 sigma^2)) * idx_m * idx_n; rowsum -> d^-1/2 ----
    if (tid < NH) {
        const int m = tid;
        const float sig = sSigma;
        const float inv2s2 = 1.0f / (2.0f * sig * sig);
        const float im = sIdx[m];
        float s = 0.f;
        for (int n = 0; n < NH; ++n) {
            float dd = (float)sL[m * NPAD + n];
            float a  = __expf(-dd * dd * inv2s2) * im * sIdx[n];
            s += a;
            sL[m * NPAD + n] = (_Float16)a;
        }
        sInv[m] = (s > 0.f) ? rsqrtf(s) : 0.0f;   // isinf -> 0 rule
    }
    __syncthreads();
    // ---- L = D^-1/2 A D^-1/2 (and zero the padding) ----
    for (int e = tid; e < NPAD * NPAD; e += 256) {
        int m = e >> 7, n = e & 127;
        float v = 0.0f;
        if (m < NH && n < NH) v = (float)sL[e] * sInv[m] * sInv[n];
        sL[e] = (_Float16)v;
    }
    __syncthreads();

    // ---- YT = (xg @ g1w + g1b)^T  (xg = feats * index), [c][m] layout ----
    for (int e = tid; e < 16 * NPAD; e += 256) {
        int c = e >> 7, m = e & 127;
        float v = 0.0f;
        if (m < NH && c < 10) {
            float acc = g1b[c];
            float im  = sIdx[m];
            #pragma unroll
            for (int k = 0; k < C3; ++k) acc += sF[m * C3 + k] * im * g1w[k * 10 + c];
            v = acc;
        }
        sYT[e] = (_Float16)v;
    }
    __syncthreads();

    // ---- h1 = elu(L @ Y) via WMMA (K = 128 = 4 steps), store h1^T ----
    {
        v8f c = {};
        for (int ks = 0; ks < 4; ++ks) {
            v16h a  = frag_rowmajor(sL + m0 * NPAD + ks * 32, NPAD, lane);
            v16h bf = frag_rowmajor(sYT + ks * 32, NPAD, lane);   // BT[n][k]
            c = __builtin_amdgcn_wmma_f32_16x16x32_f16(false, a, false, bf,
                                                       (short)0, c, false, false);
        }
        const int n  = lane & 15;
        const int mb = m0 + ((lane & 16) ? 8 : 0);
        #pragma unroll
        for (int rr = 0; rr < 8; ++rr)
            sZT[n * NPAD + (mb + rr)] = (_Float16)eluf(c[rr]);
    }
    __syncthreads();

    // ---- Y2T = (h1 @ g2w + g2b)^T, [c][m] layout ----
    for (int e = tid; e < 16 * NPAD; e += 256) {
        int c = e >> 7, m = e & 127;
        float v = 0.0f;
        if (m < NH && c < 5) {
            float acc = g2b[c];
            #pragma unroll
            for (int k = 0; k < 10; ++k)
                acc += (float)sZT[k * NPAD + m] * g2w[k * 5 + c];
            v = acc;
        }
        sYT[e] = (_Float16)v;
    }
    __syncthreads();

    // ---- h2 = elu(L @ Y2) via WMMA, result in f32 LDS ----
    {
        v8f c = {};
        for (int ks = 0; ks < 4; ++ks) {
            v16h a  = frag_rowmajor(sL + m0 * NPAD + ks * 32, NPAD, lane);
            v16h bf = frag_rowmajor(sYT + ks * 32, NPAD, lane);   // BT[n][k]
            c = __builtin_amdgcn_wmma_f32_16x16x32_f16(false, a, false, bf,
                                                       (short)0, c, false, false);
        }
        const int n  = lane & 15;
        const int mb = m0 + ((lane & 16) ? 8 : 0);
        #pragma unroll
        for (int rr = 0; rr < 8; ++rr)
            h2f[(mb + rr) * 16 + n] = eluf(c[rr]);
    }
    __syncthreads();

    // ---- pooling: pos_max / neg_max / sum over 114 nodes, 5 channels ----
    if (tid < 5) {
        const int c = tid;
        float posmax = -INFINITY, recipmin = INFINITY, hsum = 0.f;
        for (int m = 0; m < NH; ++m) {
            float v = h2f[m * 16 + c];
            posmax   = fmaxf(posmax, v);
            recipmin = fminf(recipmin, __frcp_rn(v));
            hsum    += v;
        }
        float negmax = __frcp_rn(recipmin);
        sRed[c]     = (posmax == 0.0f) ? negmax : posmax;  // hmax
        sRed[5 + c] = hsum;
    }
    __syncthreads();

    // ---- dense(10->2) + softmax ----
    if (tid == 0) {
        float l0 = db[0], l1 = db[1];
        #pragma unroll
        for (int i = 0; i < 10; ++i) {
            float f = sRed[i];
            l0 += f * dw[i * 2 + 0];
            l1 += f * dw[i * 2 + 1];
        }
        float mx = fmaxf(l0, l1);
        float e0 = __expf(l0 - mx), e1 = __expf(l1 - mx);
        float is = __frcp_rn(e0 + e1);
        out[(size_t)b * 2 + 0] = e0 * is;
        out[(size_t)b * 2 + 1] = e1 * is;
    }
}

// ============================================================
extern "C" void kernel_launch(void* const* d_in, const int* in_sizes, int n_in,
                              void* d_out, int out_size, void* d_ws, size_t ws_size,
                              hipStream_t stream)
{
    const float* x    = (const float*)d_in[0];
    const float* c1w  = (const float*)d_in[1];
    const float* c1b  = (const float*)d_in[2];
    const float* l1g  = (const float*)d_in[3];
    const float* l1b  = (const float*)d_in[4];
    const float* c2w  = (const float*)d_in[5];
    const float* c2b  = (const float*)d_in[6];
    const float* l2g  = (const float*)d_in[7];
    const float* l2b  = (const float*)d_in[8];
    const float* c3w  = (const float*)d_in[9];
    const float* c3b  = (const float*)d_in[10];
    const float* l3g  = (const float*)d_in[11];
    const float* l3b  = (const float*)d_in[12];
    const float* agw  = (const float*)d_in[13];
    const float* agb  = (const float*)d_in[14];
    const float* g1w  = (const float*)d_in[15];
    const float* g1b  = (const float*)d_in[16];
    const float* g2w  = (const float*)d_in[17];
    const float* g2b  = (const float*)d_in[18];
    const float* dw   = (const float*)d_in[19];
    const float* dB   = (const float*)d_in[20];
    const int*   rl   = (const int*)d_in[21];

    const int B = in_sizes[0] / (NH * W0);      // 1024
    float* feats = (float*)d_ws;                // B*114*20 f32 (~9.3 MB)

    gcir3_conv_stack<<<B * NH, 64, 0, stream>>>(
        x, c1w, c1b, l1g, l1b, c2w, c2b, l2g, l2b, c3w, c3b, l3g, l3b, feats);

    gcir3_graph_head<<<B, 256, 0, stream>>>(
        feats, agw, agb, g1w, g1b, g2w, g2b, dw, dB, rl, (float*)d_out);
}
